// TriangularSylvesterEdgeNetVAE_61392262529603
// MI455X (gfx1250) — compile-verified, matrix operand order checked by reading, not run on hardware
//
#include <hip/hip_runtime.h>

#define N_NODES 20000
#define N_EDGES 320000
#define BN_EPS 1e-5f

typedef __attribute__((ext_vector_type(16))) __bf16 v16bf;
typedef __attribute__((ext_vector_type(8)))  float  v8f;

static __device__ __forceinline__ unsigned short f2bf(float f) {
  unsigned int u = __float_as_uint(f);
  u += 0x7FFFu + ((u >> 16) & 1u);            // round-to-nearest-even
  return (unsigned short)(u >> 16);
}
static __device__ __forceinline__ float bf2f(unsigned short h) {
  return __uint_as_float(((unsigned int)h) << 16);
}

// ---------------------------------------------------------------- utilities
__global__ void zero_kernel(float* p, int n) {
  int i = blockIdx.x * blockDim.x + threadIdx.x;
  if (i < n) p[i] = 0.0f;
}

__global__ void count_kernel(const int* __restrict__ dstIdx, float* __restrict__ cnt) {
  int e = blockIdx.x * blockDim.x + threadIdx.x;
  if (e < N_EDGES) atomicAdd(&cnt[dstIdx[e]], 1.0f);
}

// pack W (fin x fout, row-major f32) -> Wt bf16 [foutPad][finPad] (k contiguous)
__global__ void pack_w(const float* __restrict__ W, unsigned short* __restrict__ Wt,
                       int fin, int fout, int finPad, int foutPad) {
  int idx = blockIdx.x * blockDim.x + threadIdx.x;
  if (idx >= finPad * foutPad) return;
  int nn = idx / finPad, k = idx - nn * finPad;
  float v = (nn < fout && k < fin) ? W[(size_t)k * fout + nn] : 0.0f;
  Wt[idx] = f2bf(v);
}

// concat the 6 head layers into one 64x64 (transposed bf16) + packed bias
__global__ void pack_heads(const float* muW, const float* varW, const float* dW,
                           const float* d1W, const float* d2W, const float* bW,
                           const float* muB, const float* varB, const float* dB,
                           const float* d1B, const float* d2B, const float* bB,
                           unsigned short* Wt, float* bias) {
  int idx = blockIdx.x * blockDim.x + threadIdx.x;
  if (idx >= 64 * 64) return;
  int nn = idx >> 6, k = idx & 63;
  const float *Wsrc, *Bsrc; int col, wid;
  if (nn < 2)       { Wsrc = muW;  Bsrc = muB;  col = nn;      wid = 2;  }
  else if (nn < 4)  { Wsrc = varW; Bsrc = varB; col = nn - 2;  wid = 2;  }
  else if (nn < 28) { Wsrc = dW;   Bsrc = dB;   col = nn - 4;  wid = 24; }
  else if (nn < 40) { Wsrc = d1W;  Bsrc = d1B;  col = nn - 28; wid = 12; }
  else if (nn < 52) { Wsrc = d2W;  Bsrc = d2B;  col = nn - 40; wid = 12; }
  else              { Wsrc = bW;   Bsrc = bB;   col = nn - 52; wid = 12; }
  Wt[(size_t)nn * 64 + k] = f2bf(Wsrc[(size_t)k * wid + col]);
  if (k == 0) bias[nn] = Bsrc[col];
}

__global__ void pack_bias16(const float* b4, float* b16) {
  int i = threadIdx.x;
  if (i < 16) b16[i] = (i < 4) ? b4[i] : 0.0f;
}

// ---------------------------------------------------------------- input BN
__global__ void xstats_kernel(const float* __restrict__ x, float* __restrict__ stats) {
  __shared__ float ls[8];
  if (threadIdx.x < 8) ls[threadIdx.x] = 0.0f;
  __syncthreads();
  float s[4] = {0, 0, 0, 0}, q[4] = {0, 0, 0, 0};
  for (int n = blockIdx.x * blockDim.x + threadIdx.x; n < N_NODES; n += gridDim.x * blockDim.x) {
#pragma unroll
    for (int c = 0; c < 4; ++c) { float v = x[n * 4 + c]; s[c] += v; q[c] += v * v; }
  }
#pragma unroll
  for (int c = 0; c < 4; ++c) { atomicAdd(&ls[c], s[c]); atomicAdd(&ls[4 + c], q[c]); }
  __syncthreads();
  if (threadIdx.x < 4) {
    atomicAdd(&stats[threadIdx.x], ls[threadIdx.x]);
    atomicAdd(&stats[128 + threadIdx.x], ls[4 + threadIdx.x]);
  }
}

__global__ void bn_apply_x(const float* __restrict__ x, const float* __restrict__ stats,
                           const float* __restrict__ g, const float* __restrict__ bta,
                           float* __restrict__ h0) {
  int idx = blockIdx.x * blockDim.x + threadIdx.x;
  if (idx >= N_NODES * 4) return;
  int c = idx & 3;
  float m = stats[c] * (1.0f / N_NODES);
  float v = stats[128 + c] * (1.0f / N_NODES) - m * m;
  h0[idx] = (x[idx] - m) * rsqrtf(v + BN_EPS) * g[c] + bta[c];
}

// stats(sum,sumsq) + (g,beta) -> per-column affine s,u ; BN(x) = x*s + u
__global__ void finalize_bn(const float* __restrict__ stats, const float* __restrict__ g,
                            const float* __restrict__ bta, float invcnt,
                            float* __restrict__ su, int D) {
  int c = blockIdx.x * blockDim.x + threadIdx.x;
  if (c >= D) return;
  float m = stats[c] * invcnt;
  float v = stats[128 + c] * invcnt - m * m;
  float s = g[c] * rsqrtf(v + BN_EPS);
  su[c] = s;
  su[128 + c] = bta[c] - m * s;
}

// ------------------------------------------------------------- WMMA GEMM
// M tile = 32 rows/block (two 16-row sub-tiles per wave -> each B fragment
// is reused by two v_wmma, halving B-side load traffic).
// mode 0: gather  A[e] = [h[dst], h[src]-h[dst]]  (float node feats, K=2*Din)
// mode 1: edge    A[e] = relu?(affine?(bf16 t_in))                (K=Din)
// mode 2: node    A[n] = float node feats                          (K=Din)
// Sink: fout!=null -> f32 node rows ; else bf16 edge rows (stride Dout).
// Optional column stats (sum / sumsq) for a following batchnorm fold.
__global__ __launch_bounds__(256) void gemm_tile(
    const float* __restrict__ hin, const unsigned short* __restrict__ tin,
    const int* __restrict__ srcIdx, const int* __restrict__ dstIdx,
    const unsigned short* __restrict__ Wt, const float* __restrict__ bias,
    const float* __restrict__ su, unsigned short* __restrict__ tout,
    float* __restrict__ fout, float* __restrict__ stats,
    int mode, int Din, int Kpad, int Dout, int inRelu, int outRelu) {
  __shared__ __align__(16) unsigned short Alds[32 * 256];
  __shared__ float lstat[256];
  const int mbase = blockIdx.x * 32;
  const int K2 = (mode == 0) ? (2 * Din) : Din;

  if (mode == 1 && threadIdx.x == 0 && mbase + 32 < N_EDGES)
    __builtin_prefetch(&tin[(size_t)(mbase + 32) * Din], 0, 1);

  // stage A tile (32 x Kpad) into LDS as bf16, zero-padded in K
  for (int idx = threadIdx.x; idx < 32 * Kpad; idx += 256) {
    int ml = idx / Kpad, c = idx - ml * Kpad;
    float v = 0.0f;
    if (c < K2) {
      int m = mbase + ml;
      if (mode == 0) {
        int d = dstIdx[m], s = srcIdx[m];
        if (c < Din) v = hin[(size_t)d * Din + c];
        else { int cc = c - Din; v = hin[(size_t)s * Din + cc] - hin[(size_t)d * Din + cc]; }
      } else if (mode == 1) {
        float t = bf2f(tin[(size_t)m * Din + c]);
        if (su) t = t * su[c] + su[128 + c];
        if (inRelu) t = fmaxf(t, 0.0f);
        v = t;
      } else {
        v = hin[(size_t)m * Din + c];
      }
    }
    Alds[ml * Kpad + c] = f2bf(v);
  }
  for (int i = threadIdx.x; i < 256; i += 256) lstat[i] = 0.0f;
  __syncthreads();

  const int wave = threadIdx.x >> 5;
  const int lane = threadIdx.x & 31;
  const bool active = wave < (Dout >> 4);
  const int mrow = lane & 15;
  const int half8 = (lane >> 4) << 3;
  const int ncol = (wave << 4) + mrow;       // B/D column for this lane
  v8f acc0 = {0.f, 0.f, 0.f, 0.f, 0.f, 0.f, 0.f, 0.f};
  v8f acc1 = {0.f, 0.f, 0.f, 0.f, 0.f, 0.f, 0.f, 0.f};

  if (active) {
    for (int kk = 0; kk < Kpad; kk += 32) {
      union { v16bf v; unsigned int u[8]; } a0, a1, bf;
#pragma unroll
      for (int q = 0; q < 8; ++q) {
        int k = kk + ((q < 4) ? 0 : 16) + half8 + ((q & 3) << 1); // ISA 16-bit A 16x32 layout
        bf.u[q] = *(const unsigned int*)&Wt[(size_t)ncol * Kpad + k];
        a0.u[q] = *(const unsigned int*)&Alds[mrow * Kpad + k];
        a1.u[q] = *(const unsigned int*)&Alds[(mrow + 16) * Kpad + k];
      }
      acc0 = __builtin_amdgcn_wmma_f32_16x16x32_bf16(false, a0.v, false, bf.v,
                                                     (short)0, acc0, false, false);
      acc1 = __builtin_amdgcn_wmma_f32_16x16x32_bf16(false, a1.v, false, bf.v,
                                                     (short)0, acc1, false, false);
    }
  }

  if (active) {
    float ssum = 0.f, ssq = 0.f;
    float b = bias[ncol];
#pragma unroll
    for (int r = 0; r < 8; ++r) {
      float cv0 = acc0[r] + b;
      float cv1 = acc1[r] + b;
      if (outRelu) { cv0 = fmaxf(cv0, 0.0f); cv1 = fmaxf(cv1, 0.0f); }
      int m = ((lane >> 4) << 3) + r;         // ISA f32 C/D layout
      size_t row0 = (size_t)(mbase + m);
      size_t row1 = row0 + 16;
      if (fout) { fout[row0 * Dout + ncol] = cv0; fout[row1 * Dout + ncol] = cv1; }
      else      { tout[row0 * Dout + ncol] = f2bf(cv0); tout[row1 * Dout + ncol] = f2bf(cv1); }
      ssum += cv0 + cv1; ssq += cv0 * cv0 + cv1 * cv1;
    }
    if (stats) { atomicAdd(&lstat[ncol], ssum); atomicAdd(&lstat[128 + ncol], ssq); }
  }
  __syncthreads();
  if (stats) {
    for (int c = threadIdx.x; c < Dout; c += 256) {
      atomicAdd(&stats[c], lstat[c]);
      atomicAdd(&stats[128 + c], lstat[128 + c]);
    }
  }
}

// ------------------------------------------------------------- scatter / mean
__global__ void scatter_kernel(const unsigned short* __restrict__ t,
                               const int* __restrict__ dstIdx,
                               const float* __restrict__ su,
                               float* __restrict__ sacc, int D) {
  int e = blockIdx.x * blockDim.x + threadIdx.x;
  if (e >= N_EDGES) return;
  int d = dstIdx[e];
  for (int c = 0; c < D; ++c) {
    float v = bf2f(t[(size_t)e * D + c]);
    if (su) v = v * su[c] + su[128 + c];
    atomicAdd(&sacc[(size_t)d * D + c], v);
  }
}

__global__ void div_kernel(const float* __restrict__ sacc, const float* __restrict__ cnt,
                           float* __restrict__ hout, int D) {
  int idx = blockIdx.x * blockDim.x + threadIdx.x;
  if (idx >= N_NODES * D) return;
  int n = idx / D;
  hout[idx] = sacc[idx] / fmaxf(cnt[n], 1.0f);
}

__global__ void xdec_kernel(const float* __restrict__ sacc16, const float* __restrict__ cnt,
                            float* __restrict__ out) {
  int idx = blockIdx.x * blockDim.x + threadIdx.x;
  if (idx >= N_NODES * 4) return;
  int n = idx >> 2, c = idx & 3;
  out[idx] = sacc16[(size_t)n * 16 + c] / fmaxf(cnt[n], 1.0f);
}

// ------------------------------------------------------------- flow chain
// heads row layout: [0:2]=mu [2:4]=logvar [4:28]=amor_d [28:40]=d1 [40:52]=d2 [52:64]=b
__global__ void flow_kernel(const float* __restrict__ heads, const float* __restrict__ eps,
                            float* out_mu, float* out_lv, float* out_ld,
                            float* out_z0, float* out_zk, float* zkbuf) {
  int n = blockIdx.x * blockDim.x + threadIdx.x;
  if (n >= N_NODES) return;
  const float* hh = heads + (size_t)n * 64;
  float mu0 = hh[0], mu1 = hh[1];
  float lv0 = hh[2], lv1 = hh[3];
  float z0a = mu0 + eps[2 * n] * expf(0.5f * lv0);
  float z0b = mu1 + eps[2 * n + 1] * expf(0.5f * lv1);
  float za = z0a, zb = z0b, logdet = 0.0f;
#pragma unroll
  for (int k = 0; k < 6; ++k) {
    float r101 = hh[4 + 6 + k];            // full_d[0][1][k]
    float r201 = hh[4 + 12 + k];           // full_d[1][0][k] (transposed)
    float r100 = tanhf(hh[28 + k]);
    float r111 = tanhf(hh[34 + k]);
    float r200 = tanhf(hh[40 + k]);
    float r211 = tanhf(hh[46 + k]);
    float b0 = hh[52 + k], b1 = hh[58 + k];
    bool perm = (k & 1);
    float zp0 = perm ? zb : za;
    float zp1 = perm ? za : zb;
    float q0 = zp0 * r200 + zp1 * r201 + b0;
    float q1 = zp1 * r211 + b1;
    float h0 = tanhf(q0), h1 = tanhf(q1);
    float zn0 = h0 * r100 + h1 * r101;
    float zn1 = h1 * r111;
    if (perm) { float t = zn0; zn0 = zn1; zn1 = t; }
    za += zn0; zb += zn1;
    float dj0 = (1.0f - h0 * h0) * r100 * r200 + 1.0f;
    float dj1 = (1.0f - h1 * h1) * r111 * r211 + 1.0f;
    logdet += logf(fabsf(dj0)) + logf(fabsf(dj1));
  }
  out_mu[2 * n] = mu0; out_mu[2 * n + 1] = mu1;
  out_lv[2 * n] = lv0; out_lv[2 * n + 1] = lv1;
  out_ld[n] = logdet;
  out_z0[2 * n] = z0a; out_z0[2 * n + 1] = z0b;
  out_zk[2 * n] = za;  out_zk[2 * n + 1] = zb;
  zkbuf[2 * n] = za;   zkbuf[2 * n + 1] = zb;
}

// ---------------------------------------------------------------- host
static inline int cdiv(int a, int b) { return (a + b - 1) / b; }

extern "C" void kernel_launch(void* const* d_in, const int* in_sizes, int n_in,
                              void* d_out, int out_size, void* d_ws, size_t ws_size,
                              hipStream_t stream) {
  (void)in_sizes; (void)n_in; (void)out_size; (void)ws_size;
  const float* x   = (const float*)d_in[0];
  const int*   ei  = (const int*)d_in[1];
  const int*   src = ei;
  const int*   dst = ei + N_EDGES;
  const float* eps = (const float*)d_in[2];
#define IN_F(i) ((const float*)d_in[i])
  // jax sorted-dict flatten order (b<w, bta<g, bn*<l*)
  const float *amorB_b = IN_F(3),  *amorB_w = IN_F(4);
  const float *amorD_b = IN_F(5),  *amorD_w = IN_F(6);
  const float *amorD1_b = IN_F(7), *amorD1_w = IN_F(8);
  const float *amorD2_b = IN_F(9), *amorD2_w = IN_F(10);
  const float *bnin_bta = IN_F(11), *bnin_g = IN_F(12);
  const float *d1bn1_bta = IN_F(13), *d1bn1_g = IN_F(14);
  const float *d1bn2_bta = IN_F(15), *d1bn2_g = IN_F(16);
  const float *d1l1_b = IN_F(17), *d1l1_w = IN_F(18);
  const float *d1l2_b = IN_F(19), *d1l2_w = IN_F(20);
  const float *d1l3_b = IN_F(21), *d1l3_w = IN_F(22);
  const float *d2bn1_bta = IN_F(23), *d2bn1_g = IN_F(24);
  const float *d2bn2_bta = IN_F(25), *d2bn2_g = IN_F(26);
  const float *d2l1_b = IN_F(27), *d2l1_w = IN_F(28);
  const float *d2l2_b = IN_F(29), *d2l2_w = IN_F(30);
  const float *d2l3_b = IN_F(31), *d2l3_w = IN_F(32);
  const float *e1bn1_bta = IN_F(33), *e1bn1_g = IN_F(34);
  const float *e1bn2_bta = IN_F(35), *e1bn2_g = IN_F(36);
  const float *e1bn3_bta = IN_F(37), *e1bn3_g = IN_F(38);
  const float *e1l1_b = IN_F(39), *e1l1_w = IN_F(40);
  const float *e1l2_b = IN_F(41), *e1l2_w = IN_F(42);
  const float *e1l3_b = IN_F(43), *e1l3_w = IN_F(44);
  const float *e2bn1_bta = IN_F(45), *e2bn1_g = IN_F(46);
  const float *e2bn2_bta = IN_F(47), *e2bn2_g = IN_F(48);
  const float *e2l1_b = IN_F(49), *e2l1_w = IN_F(50);
  const float *e2l2_b = IN_F(51), *e2l2_w = IN_F(52);
  const float *mu_b = IN_F(53), *mu_w = IN_F(54);
  const float *var_b = IN_F(55), *var_w = IN_F(56);

  float* out_xdec = (float*)d_out;
  float* out_mu   = out_xdec + (size_t)N_NODES * 4;
  float* out_lv   = out_mu + (size_t)N_NODES * 2;
  float* out_ld   = out_lv + (size_t)N_NODES * 2;
  float* out_z0   = out_ld + N_NODES;
  float* out_zk   = out_z0 + (size_t)N_NODES * 2;

  char* ws = (char*)d_ws;
  size_t off = 0;
  auto take = [&](size_t bytes) -> void* {
    void* p = ws + off;
    off += (bytes + 255) & ~(size_t)255;
    return p;
  };
  float* cnt     = (float*)take((size_t)N_NODES * 4);
  float* stats   = (float*)take(256 * 4);
  float* suA     = (float*)take(256 * 4);
  float* suB     = (float*)take(256 * 4);
  float* h0      = (float*)take((size_t)N_NODES * 4 * 4);
  float* hA      = (float*)take((size_t)N_NODES * 128 * 4);   // h1 (64) then hd (128)
  float* hB      = (float*)take((size_t)N_NODES * 64 * 4);    // h2
  float* zkbuf   = (float*)take((size_t)N_NODES * 2 * 4);
  float* headsB  = (float*)take((size_t)N_NODES * 64 * 4);
  float* sacc    = (float*)take((size_t)N_NODES * 128 * 4);
  unsigned short* tA = (unsigned short*)take((size_t)N_EDGES * 128 * 2);
  unsigned short* tB = (unsigned short*)take((size_t)N_EDGES * 128 * 2);
  unsigned short* w_e1l1 = (unsigned short*)take(128 * 32 * 2);
  unsigned short* w_e1l2 = (unsigned short*)take(128 * 128 * 2);
  unsigned short* w_e1l3 = (unsigned short*)take(64 * 128 * 2);
  unsigned short* w_e2l1 = (unsigned short*)take(64 * 128 * 2);
  unsigned short* w_e2l2 = (unsigned short*)take(64 * 64 * 2);
  unsigned short* w_hd   = (unsigned short*)take(64 * 64 * 2);
  unsigned short* w_d1l1 = (unsigned short*)take(64 * 32 * 2);
  unsigned short* w_d1l2 = (unsigned short*)take(64 * 64 * 2);
  unsigned short* w_d1l3 = (unsigned short*)take(128 * 64 * 2);
  unsigned short* w_d2l1 = (unsigned short*)take(128 * 256 * 2);
  unsigned short* w_d2l2 = (unsigned short*)take(128 * 128 * 2);
  unsigned short* w_d2l3 = (unsigned short*)take(16 * 128 * 2);
  float* headsBias = (float*)take(64 * 4);
  float* bias16    = (float*)take(16 * 4);

  const int ET = N_EDGES / 32, NT = N_NODES / 32;   // 10000 / 625, both exact
#define PACK(Wsrc, Wdst, fi, fo, fip, fop) \
  pack_w<<<cdiv((fip) * (fop), 256), 256, 0, stream>>>(Wsrc, Wdst, fi, fo, fip, fop)
  PACK(e1l1_w, w_e1l1, 8, 128, 32, 128);
  PACK(e1l2_w, w_e1l2, 128, 128, 128, 128);
  PACK(e1l3_w, w_e1l3, 128, 64, 128, 64);
  PACK(e2l1_w, w_e2l1, 128, 64, 128, 64);
  PACK(e2l2_w, w_e2l2, 64, 64, 64, 64);
  PACK(d1l1_w, w_d1l1, 4, 64, 32, 64);
  PACK(d1l2_w, w_d1l2, 64, 64, 64, 64);
  PACK(d1l3_w, w_d1l3, 64, 128, 64, 128);
  PACK(d2l1_w, w_d2l1, 256, 128, 256, 128);
  PACK(d2l2_w, w_d2l2, 128, 128, 128, 128);
  PACK(d2l3_w, w_d2l3, 128, 4, 128, 16);
  pack_heads<<<16, 256, 0, stream>>>(mu_w, var_w, amorD_w, amorD1_w, amorD2_w, amorB_w,
                                     mu_b, var_b, amorD_b, amorD1_b, amorD2_b, amorB_b,
                                     w_hd, headsBias);
  pack_bias16<<<1, 16, 0, stream>>>(d2l3_b, bias16);

  zero_kernel<<<cdiv(N_NODES, 256), 256, 0, stream>>>(cnt, N_NODES);
  count_kernel<<<cdiv(N_EDGES, 256), 256, 0, stream>>>(dst, cnt);

  // input batchnorm
  zero_kernel<<<1, 256, 0, stream>>>(stats, 256);
  xstats_kernel<<<128, 256, 0, stream>>>(x, stats);
  bn_apply_x<<<cdiv(N_NODES * 4, 256), 256, 0, stream>>>(x, stats, bnin_g, bnin_bta, h0);

  const float invE = 1.0f / (float)N_EDGES;
#define ZS() zero_kernel<<<1, 256, 0, stream>>>(stats, 256)

  // ---- enc1: 8->128->128->64 (BN after relu each layer) ----
  ZS();
  gemm_tile<<<ET, 256, 0, stream>>>(h0, nullptr, src, dst, w_e1l1, e1l1_b, nullptr,
                                    tA, nullptr, stats, 0, 4, 32, 128, 0, 1);
  finalize_bn<<<1, 128, 0, stream>>>(stats, e1bn1_g, e1bn1_bta, invE, suA, 128);
  ZS();
  gemm_tile<<<ET, 256, 0, stream>>>(nullptr, tA, src, dst, w_e1l2, e1l2_b, suA,
                                    tB, nullptr, stats, 1, 128, 128, 128, 0, 1);
  finalize_bn<<<1, 128, 0, stream>>>(stats, e1bn2_g, e1bn2_bta, invE, suB, 128);
  ZS();
  gemm_tile<<<ET, 256, 0, stream>>>(nullptr, tB, src, dst, w_e1l3, e1l3_b, suB,
                                    tA, nullptr, stats, 1, 128, 128, 64, 0, 1);
  finalize_bn<<<1, 128, 0, stream>>>(stats, e1bn3_g, e1bn3_bta, invE, suA, 64);
  zero_kernel<<<cdiv(N_NODES * 64, 256), 256, 0, stream>>>(sacc, N_NODES * 64);
  scatter_kernel<<<cdiv(N_EDGES, 256), 256, 0, stream>>>(tA, dst, suA, sacc, 64);
  div_kernel<<<cdiv(N_NODES * 64, 256), 256, 0, stream>>>(sacc, cnt, hA, 64);

  // ---- enc2: 128->64->64 ----
  ZS();
  gemm_tile<<<ET, 256, 0, stream>>>(hA, nullptr, src, dst, w_e2l1, e2l1_b, nullptr,
                                    tA, nullptr, stats, 0, 64, 128, 64, 0, 1);
  finalize_bn<<<1, 128, 0, stream>>>(stats, e2bn1_g, e2bn1_bta, invE, suA, 64);
  ZS();
  gemm_tile<<<ET, 256, 0, stream>>>(nullptr, tA, src, dst, w_e2l2, e2l2_b, suA,
                                    tB, nullptr, stats, 1, 64, 64, 64, 0, 1);
  finalize_bn<<<1, 128, 0, stream>>>(stats, e2bn2_g, e2bn2_bta, invE, suB, 64);
  zero_kernel<<<cdiv(N_NODES * 64, 256), 256, 0, stream>>>(sacc, N_NODES * 64);
  scatter_kernel<<<cdiv(N_EDGES, 256), 256, 0, stream>>>(tB, dst, suB, sacc, 64);
  div_kernel<<<cdiv(N_NODES * 64, 256), 256, 0, stream>>>(sacc, cnt, hB, 64);

  // ---- heads (mu|var|amor_d|diag1|diag2|amor_b packed as one 64x64) + flow ----
  gemm_tile<<<NT, 256, 0, stream>>>(hB, nullptr, src, dst, w_hd, headsBias, nullptr,
                                    nullptr, headsB, nullptr, 2, 64, 64, 64, 0, 0);
  flow_kernel<<<cdiv(N_NODES, 256), 256, 0, stream>>>(headsB, eps, out_mu, out_lv,
                                                      out_ld, out_z0, out_zk, zkbuf);

  // ---- dec1: 4->64 (relu), ->64 (BN then relu), ->128 (relu then BN) ----
  gemm_tile<<<ET, 256, 0, stream>>>(zkbuf, nullptr, src, dst, w_d1l1, d1l1_b, nullptr,
                                    tA, nullptr, nullptr, 0, 2, 32, 64, 0, 1);
  ZS();
  gemm_tile<<<ET, 256, 0, stream>>>(nullptr, tA, src, dst, w_d1l2, d1l2_b, nullptr,
                                    tB, nullptr, stats, 1, 64, 64, 64, 0, 0);
  finalize_bn<<<1, 128, 0, stream>>>(stats, d1bn1_g, d1bn1_bta, invE, suA, 64);
  ZS();
  gemm_tile<<<ET, 256, 0, stream>>>(nullptr, tB, src, dst, w_d1l3, d1l3_b, suA,
                                    tA, nullptr, stats, 1, 64, 64, 128, 1, 1);
  finalize_bn<<<1, 128, 0, stream>>>(stats, d1bn2_g, d1bn2_bta, invE, suB, 128);
  zero_kernel<<<cdiv(N_NODES * 128, 256), 256, 0, stream>>>(sacc, N_NODES * 128);
  scatter_kernel<<<cdiv(N_EDGES, 256), 256, 0, stream>>>(tA, dst, suB, sacc, 128);
  div_kernel<<<cdiv(N_NODES * 128, 256), 256, 0, stream>>>(sacc, cnt, hA, 128);

  // ---- dec2: 256->128->128->4 ----
  ZS();
  gemm_tile<<<ET, 256, 0, stream>>>(hA, nullptr, src, dst, w_d2l1, d2l1_b, nullptr,
                                    tA, nullptr, stats, 0, 128, 256, 128, 0, 1);
  finalize_bn<<<1, 128, 0, stream>>>(stats, d2bn1_g, d2bn1_bta, invE, suA, 128);
  ZS();
  gemm_tile<<<ET, 256, 0, stream>>>(nullptr, tA, src, dst, w_d2l2, d2l2_b, suA,
                                    tB, nullptr, stats, 1, 128, 128, 128, 0, 1);
  finalize_bn<<<1, 128, 0, stream>>>(stats, d2bn2_g, d2bn2_bta, invE, suB, 128);
  gemm_tile<<<ET, 256, 0, stream>>>(nullptr, tB, src, dst, w_d2l3, bias16, suB,
                                    tA, nullptr, nullptr, 1, 128, 128, 16, 0, 0);
  zero_kernel<<<cdiv(N_NODES * 16, 256), 256, 0, stream>>>(sacc, N_NODES * 16);
  scatter_kernel<<<cdiv(N_EDGES, 256), 256, 0, stream>>>(tA, dst, nullptr, sacc, 16);
  xdec_kernel<<<cdiv(N_NODES * 4, 256), 256, 0, stream>>>(sacc, cnt, out_xdec);
}
